// Expert_int_90950227460275
// MI455X (gfx1250) — 8-bit matrix operand order repaired and checked by reading, compiled, not run on hardware
//
#include <hip/hip_runtime.h>


typedef int v8i __attribute__((ext_vector_type(8)));

#define T_DIM 8192
#define D_DIM 2048
#define F_DIM 1408
#define QSH   23

#ifndef __has_builtin
#define __has_builtin(x) 0
#endif

#if __has_builtin(__builtin_amdgcn_global_load_async_to_lds_b64)
#define HAVE_ASYNC_LDS 1
typedef int v2i_vs __attribute__((__vector_size__(2 * sizeof(int))));
typedef __attribute__((address_space(1))) v2i_vs gas_v2i;
typedef __attribute__((address_space(3))) v2i_vs las_v2i;
#define ASYNC_B64(gp, lp)                                                      \
  __builtin_amdgcn_global_load_async_to_lds_b64((gas_v2i*)(gp),                \
                                                (las_v2i*)(lp), 0, 0)
#if __has_builtin(__builtin_amdgcn_s_wait_asynccnt)
#define WAIT_ASYNC0() __builtin_amdgcn_s_wait_asynccnt(0)
#else
#define WAIT_ASYNC0() asm volatile("s_wait_asynccnt 0x0" ::: "memory")
#endif
#else
#define HAVE_ASYNC_LDS 0
#define ASYNC_B64(gp, lp) (*(long long*)(lp) = *(const long long*)(gp))
#define WAIT_ASYNC0() ((void)0)
#endif

__device__ __forceinline__ v8i wmma_iu8(v8i a, v8i b, v8i c) {
  return __builtin_amdgcn_wmma_i32_16x16x64_iu8(true, a, true, b, c, false, false);
}

__device__ __forceinline__ v8i load_b_frag(const char* __restrict__ p) {
  const int4 q0 = *(const int4*)(p);
  const int4 q1 = *(const int4*)(p + 32);
  v8i b;
  b[0] = q0.x; b[1] = q0.y; b[2] = q0.z; b[3] = q0.w;
  b[4] = q1.x; b[5] = q1.y; b[6] = q1.z; b[7] = q1.w;
  return b;
}

__device__ __forceinline__ v8i load_a_frag_lds(const char* bp) {
  const int2 q0 = *(const int2*)(bp);
  const int2 q1 = *(const int2*)(bp + 8);
  const int2 q2 = *(const int2*)(bp + 32);
  const int2 q3 = *(const int2*)(bp + 40);
  v8i a;
  a[0] = q0.x; a[1] = q0.y; a[2] = q1.x; a[3] = q1.y;
  a[4] = q2.x; a[5] = q2.y; a[6] = q3.x; a[7] = q3.y;
  return a;
}


__global__ void pack_w_kernel(const int* __restrict__ src, char* __restrict__ dst, int n) {
  for (int i = blockIdx.x * blockDim.x + threadIdx.x; i < n; i += gridDim.x * blockDim.x)
    dst[i] = (char)src[i];
}

__global__ void pack_x_kernel(const int* __restrict__ x, char* __restrict__ l0,
                              char* __restrict__ l1, char* __restrict__ l2, int n) {
  for (int i = blockIdx.x * blockDim.x + threadIdx.x; i < n; i += gridDim.x * blockDim.x) {
    const int v  = x[i];
    const int b0 = (int)(signed char)(v & 0xff);
    const int t1 = (v - b0) >> 8;
    const int b1 = (int)(signed char)(t1 & 0xff);
    const int b2 = (t1 - b1) >> 8;
    l0[i] = (char)b0; l1[i] = (char)b1; l2[i] = (char)b2;
  }
}


__global__ __launch_bounds__(256) void gate_kernel(
    const char* __restrict__ l0, const char* __restrict__ l1, const char* __restrict__ l2,
    const char* __restrict__ w1b, const char* __restrict__ w3b,
    const int* __restrict__ sc1p, const int* __restrict__ sc3p,
    char* __restrict__ gate)
{
  __shared__ __align__(16) char smA[2][3][16 * 64];

  const int tid  = threadIdx.x;
  const int lane = tid & 31;
  const int wave = tid >> 5;
  const int m16  = lane & 15;
  const int kh   = lane >> 4;
  const int mbase = blockIdx.y << 4;
  const int nbase = (blockIdx.x << 7) + (wave << 4);

  const bool stager = (tid < 128);
  const int cr = (tid & 127) >> 3;
  const int cc = (tid & 7) << 3;
  const size_t grow = (size_t)(mbase + cr) * D_DIM + cc;
  const int lds_off = (cr << 6) + cc;

  v8i acc1[3], acc3[3];
#pragma unroll
  for (int l = 0; l < 3; ++l) { acc1[l] = (v8i)0; acc3[l] = (v8i)0; }

  if (stager) {
    ASYNC_B64(l0 + grow, &smA[0][0][lds_off]);
    ASYNC_B64(l1 + grow, &smA[0][1][lds_off]);
    ASYNC_B64(l2 + grow, &smA[0][2][lds_off]);
  }

  const int f = nbase + m16;
  const int NSTEP = D_DIM / 64;
  for (int ks = 0; ks < NSTEP; ++ks) {
    const int k0  = ks << 6;
    const int cur = ks & 1;

    WAIT_ASYNC0();
    __syncthreads();

    if (stager && ks + 1 < NSTEP) {
      const size_t g = grow + (size_t)(k0 + 64);
      ASYNC_B64(l0 + g, &smA[cur ^ 1][0][lds_off]);
      ASYNC_B64(l1 + g, &smA[cur ^ 1][1][lds_off]);
      ASYNC_B64(l2 + g, &smA[cur ^ 1][2][lds_off]);
    }

    const int arow = (m16 << 6) + (kh << 4);
    const v8i a0 = load_a_frag_lds(&smA[cur][0][arow]);
    const v8i a1 = load_a_frag_lds(&smA[cur][1][arow]);
    const v8i a2 = load_a_frag_lds(&smA[cur][2][arow]);

    const char* p1 = w1b + f * D_DIM + k0 + (kh << 4);
    const char* p3 = w3b + f * D_DIM + k0 + (kh << 4);
    __builtin_prefetch(p1 + 64, 0, 0);
    __builtin_prefetch(p3 + 64, 0, 0);
    const v8i b1 = load_b_frag(p1);
    const v8i b3 = load_b_frag(p3);
    acc1[0] = wmma_iu8(a0, b1, acc1[0]);
    acc1[1] = wmma_iu8(a1, b1, acc1[1]);
    acc1[2] = wmma_iu8(a2, b1, acc1[2]);
    acc3[0] = wmma_iu8(a0, b3, acc3[0]);
    acc3[1] = wmma_iu8(a1, b3, acc3[1]);
    acc3[2] = wmma_iu8(a2, b3, acc3[2]);
  }

  const int s1 = *sc1p, s3 = *sc3p;
  const int col = nbase + m16;
#pragma unroll
  for (int v = 0; v < 8; ++v) {
    long long r1 = (long long)acc1[0][v] + ((long long)acc1[1][v] << 8)
                 + ((long long)acc1[2][v] << 16);
    long long r2 = (long long)acc3[0][v] + ((long long)acc3[1][v] << 8)
                 + ((long long)acc3[2][v] << 16);
    r1 >>= s1;
    r2 >>= s3;
    const double xf = (double)r1 * (1.0 / 8388608.0);
    const double y  = xf / (1.0 + exp(-xf));
    const long long s1q = llrint(y * 8388608.0);
    long long g = (s1q * r2) >> QSH;
    g = g < -128 ? -128 : (g > 127 ? 127 : g);
    const int row = mbase + v + (kh << 3);
    gate[row * F_DIM + col] = (char)g;
  }
}


__global__ __launch_bounds__(128) void out_kernel(
    const char* __restrict__ gate, const char* __restrict__ w2b,
    const int* __restrict__ sc2p, float* __restrict__ out)
{
  __shared__ __align__(16) char smA[2][16 * 64];

  const int tid  = threadIdx.x;
  const int lane = tid & 31;
  const int wave = tid >> 5;
  const int m16  = lane & 15;
  const int kh   = lane >> 4;
  const int mbase = blockIdx.y << 4;
  const int nbase = (blockIdx.x << 8) + (wave << 6);

  const int cr = tid >> 3;
  const int cc = (tid & 7) << 3;
  const size_t grow = (size_t)(mbase + cr) * F_DIM + cc;
  const int lds_off = (cr << 6) + cc;

  v8i acc[4];
#pragma unroll
  for (int t = 0; t < 4; ++t) acc[t] = (v8i)0;

  ASYNC_B64(gate + grow, &smA[0][lds_off]);

  const int NSTEP = F_DIM / 64;
  for (int ks = 0; ks < NSTEP; ++ks) {
    const int k0  = ks << 6;
    const int cur = ks & 1;

    WAIT_ASYNC0();
    __syncthreads();

    if (ks + 1 < NSTEP)
      ASYNC_B64(gate + grow + (size_t)(k0 + 64), &smA[cur ^ 1][lds_off]);

    const v8i a = load_a_frag_lds(&smA[cur][(m16 << 6) + (kh << 4)]);

#pragma unroll
    for (int t = 0; t < 4; ++t) {
      const int f = nbase + (t << 4) + m16;
      const char* p = w2b + f * F_DIM + k0 + (kh << 4);
      __builtin_prefetch(p + 64, 0, 0);
      const v8i b = load_b_frag(p);
      acc[t] = wmma_iu8(a, b, acc[t]);
    }
  }

  const int sc = *sc2p;
#pragma unroll
  for (int t = 0; t < 4; ++t) {
    const int col = nbase + (t << 4) + m16;
#pragma unroll
    for (int v = 0; v < 8; ++v) {
      const int q = acc[t][v] >> sc;
      out[(mbase + v + (kh << 3)) * D_DIM + col] = (float)q;
    }
  }
}


extern "C" void kernel_launch(void* const* d_in, const int* in_sizes, int n_in,
                              void* d_out, int out_size, void* d_ws, size_t ws_size,
                              hipStream_t stream) {
  (void)in_sizes; (void)n_in; (void)out_size; (void)ws_size;

  const int* x   = (const int*)d_in[0];
  const int* w1  = (const int*)d_in[1];
  const int* w2  = (const int*)d_in[2];
  const int* w3  = (const int*)d_in[3];
  const int* sc1 = (const int*)d_in[4];
  const int* sc2 = (const int*)d_in[5];
  const int* sc3 = (const int*)d_in[6];
  float* out = (float*)d_out;

  char* ws = (char*)d_ws;
  size_t off = 0;
  auto carve = [&](size_t bytes) -> char* {
    char* p = ws + off;
    off += (bytes + 255) & ~(size_t)255;
    return p;
  };

  const size_t TD = (size_t)T_DIM * D_DIM;
  const size_t FD = (size_t)F_DIM * D_DIM;
  const size_t TF = (size_t)T_DIM * F_DIM;

  char* l0   = carve(TD);
  char* l1   = carve(TD);
  char* l2   = carve(TD);
  char* w1b  = carve(FD);
  char* w3b  = carve(FD);
  char* w2b  = carve(FD);
  char* gate = carve(TF);

  pack_x_kernel<<<4096, 256, 0, stream>>>(x, l0, l1, l2, (int)TD);
  pack_w_kernel<<<2048, 256, 0, stream>>>(w1, w1b, (int)FD);
  pack_w_kernel<<<2048, 256, 0, stream>>>(w3, w3b, (int)FD);
  pack_w_kernel<<<2048, 256, 0, stream>>>(w2, w2b, (int)FD);

  gate_kernel<<<dim3(F_DIM / 128, T_DIM / 16), 256, 0, stream>>>(
      l0, l1, l2, w1b, w3b, sc1, sc3, gate);

  out_kernel<<<dim3(D_DIM / 256, T_DIM / 16), 128, 0, stream>>>(
      gate, w2b, sc2, out);
}
